// LinearAttention_30219389894970
// MI455X (gfx1250) — matrix operand order repaired and checked by reading, not compiled, not run on hardware
//
#include <hip/hip_runtime.h>

// ---------------------------------------------------------------------------
// CDNA5 (gfx1250) linear-attention pipeline, bf16 WMMA everywhere.
// A operands row-major M x K; B operands stored transposed N x K.
// Heavy GEMMs stage 128x32 A/B slabs through double-buffered LDS:
//   global->regs (next slab) | ds_load frags + 8 wmma (cur slab) |
//   regs->LDS (next slab, load-wait hidden behind wmma) | split barrier.
// ---------------------------------------------------------------------------

typedef __bf16 bf16_t;
typedef __attribute__((ext_vector_type(16))) __bf16 v16bf;
typedef __attribute__((ext_vector_type(8)))  float  v8f;

union FragU  { v16bf v; uint4 q[2]; };
union Pack8  { bf16_t e[8]; uint4 u; };
union Pack4  { bf16_t e[4]; uint2 u; };

#define LDS_PAD 40   // 32 K-elems + 8 pad: 80B row pitch breaks bank conflicts

__device__ __forceinline__ v8f zero8() {
  v8f z = {0.f, 0.f, 0.f, 0.f, 0.f, 0.f, 0.f, 0.f};
  return z;
}

// A fragment: 16x32 bf16 tile, source row-major, leading dim `ld` (elements).
// ISA layout: lanes 0-15 -> M=lane, K {0..7} in v0-3 and {16..23} in v4-7;
//             lanes 16-31 -> M=lane-16, K {8..15} and {24..31}.
__device__ __forceinline__ v16bf load_a_frag(const bf16_t* base, int ld) {
  const int lane = threadIdx.x & 31;
  const bf16_t* p = base + (size_t)(lane & 15) * ld + ((lane >> 4) << 3);
  FragU f;
  f.q[0] = *(const uint4*)(p);        // K = kb .. kb+7
  f.q[1] = *(const uint4*)(p + 16);   // K = kb+16 .. kb+23
  return f.v;
}

// B fragment: 32x16 bf16 tile, source is B^T (N x K row-major), leading dim ld.
// ISA layout: lanes 0-15 -> N=lane, K {0..7} and {16..23}; lanes 16-31 -> K {8..15} and {24..31}.
__device__ __forceinline__ v16bf load_b_frag(const bf16_t* baseT, int ld) {
  const int lane = threadIdx.x & 31;
  const bf16_t* p = baseT + (size_t)(lane & 15) * ld + ((lane >> 4) << 3);
  FragU f;
  f.q[0] = *(const uint4*)(p);        // K = kb .. kb+7
  f.q[1] = *(const uint4*)(p + 16);   // K = kb+16 .. kb+23
  return f.v;
}

__device__ __forceinline__ v8f wmma_bf16(v16bf a, v16bf b, v8f c) {
  return __builtin_amdgcn_wmma_f32_16x16x32_bf16(false, a, false, b,
                                                 (short)0, c, false, false);
}

// ---- 128x32 slab movement: each of 256 threads moves 16 consecutive bf16 ---
__device__ __forceinline__ void slab_load(uint4& r0, uint4& r1,
                                          const bf16_t* __restrict__ src,
                                          size_t ld) {
  const int t   = threadIdx.x;
  const int row = t >> 1;
  const int col = (t & 1) << 4;
  const bf16_t* p = src + (size_t)row * ld + col;
  r0 = *(const uint4*)(p);
  r1 = *(const uint4*)(p + 8);
}

__device__ __forceinline__ void slab_store(bf16_t* dst,
                                           const uint4& r0, const uint4& r1) {
  const int t   = threadIdx.x;
  const int row = t >> 1;
  const int col = (t & 1) << 4;
  bf16_t* p = dst + row * LDS_PAD + col;
  *(uint4*)(p)     = r0;
  *(uint4*)(p + 8) = r1;
}

// ---------------------------------------------------------------------------
// Prep kernels
// ---------------------------------------------------------------------------

__global__ __launch_bounds__(256) void cvt_f32_bf16_x4(const float* __restrict__ x,
                                                       bf16_t* __restrict__ y,
                                                       long n4) {
  const long i = (long)blockIdx.x * 256 + threadIdx.x;
  if (i >= n4) return;
  const float4 f = ((const float4*)x)[i];
  Pack4 p;
  p.e[0] = (bf16_t)f.x; p.e[1] = (bf16_t)f.y;
  p.e[2] = (bf16_t)f.z; p.e[3] = (bf16_t)f.w;
  ((uint2*)y)[i] = p.u;
}

// wt[n*K + k] = (bf16) w[k*N + n]   (w is K rows x N cols, row-major f32)
__global__ __launch_bounds__(256) void transpose_f32_bf16(const float* __restrict__ w,
                                                          bf16_t* __restrict__ wt,
                                                          int K, int N) {
  const long idx = (long)blockIdx.x * 256 + threadIdx.x;
  if (idx >= (long)K * N) return;
  const int n = (int)(idx / K);
  const int k = (int)(idx % K);
  wt[idx] = (bf16_t)w[(long)k * N + n];
}

// ---------------------------------------------------------------------------
// GEMM1: qkv = x @ w_qkv.  M=32768, N=3072, K=1024.
// WG = 8 waves (4 m-blocks x 2 n-blocks), wave tile 32x64, WG tile 128x128.
// LDS double-buffered 128x32 slabs for both A and B.
// Epilogue: cols [0,1024)    -> QRAW f32 row-major
//           cols [1024,2048) -> KT  bf16 transposed [(b*16+h)*64+dd][n]
//           cols [2048,3072) -> VT  bf16 transposed likewise
// ---------------------------------------------------------------------------
__global__ __launch_bounds__(256, 1) void gemm_qkv(const bf16_t* __restrict__ XB,
                                                   const bf16_t* __restrict__ WQKVT,
                                                   float* __restrict__ QRAW,
                                                   bf16_t* __restrict__ KT,
                                                   bf16_t* __restrict__ VT) {
  __shared__ bf16_t Asl[2][128 * LDS_PAD];
  __shared__ bf16_t Bsl[2][128 * LDS_PAD];
  const int lane = threadIdx.x & 31;
  const int wave = threadIdx.x >> 5;
  const int wm = wave >> 1;            // 0..3
  const int wn = wave & 1;             // 0..1
  const int row0 = blockIdx.y * 128;
  const int col0 = blockIdx.x * 128;
  v8f acc[8];
#pragma unroll
  for (int i = 0; i < 8; ++i) acc[i] = zero8();

  const bf16_t* Ag = XB + (size_t)row0 * 1024;
  const bf16_t* Bg = WQKVT + (size_t)col0 * 1024;

  uint4 ra0, ra1, rb0, rb1;
  slab_load(ra0, ra1, Ag, 1024);
  slab_load(rb0, rb1, Bg, 1024);
  slab_store(Asl[0], ra0, ra1);
  slab_store(Bsl[0], rb0, rb1);
  __syncthreads();

  for (int kk = 0; kk < 1024; kk += 32) {
    const int cur = (kk >> 5) & 1;
    const bool more = (kk + 32) < 1024;
    if (more) {                         // issue next slab's global loads now
      slab_load(ra0, ra1, Ag + kk + 32, 1024);
      slab_load(rb0, rb1, Bg + kk + 32, 1024);
    }
    const bf16_t* Ab = Asl[cur] + (wm * 32) * LDS_PAD;
    const bf16_t* Bb = Bsl[cur] + (wn * 64) * LDS_PAD;
    const v16bf a0 = load_a_frag(Ab, LDS_PAD);
    const v16bf a1 = load_a_frag(Ab + 16 * LDS_PAD, LDS_PAD);
    v16bf bf[4];
#pragma unroll
    for (int t = 0; t < 4; ++t)         // all ds_loads issue up front
      bf[t] = load_b_frag(Bb + 16 * t * LDS_PAD, LDS_PAD);
#pragma unroll
    for (int t = 0; t < 4; ++t) {
      acc[t]     = wmma_bf16(a0, bf[t], acc[t]);
      acc[4 + t] = wmma_bf16(a1, bf[t], acc[4 + t]);
    }
    if (more) {                         // load-wait lands here, behind wmmas
      slab_store(Asl[cur ^ 1], ra0, ra1);
      slab_store(Bsl[cur ^ 1], rb0, rb1);
    }
    __syncthreads();
  }

  const int mbase = (lane >> 4) << 3;   // 0 or 8
  const int nn    = lane & 15;          // N within 16-wide tile
#pragma unroll
  for (int mb = 0; mb < 2; ++mb) {
    const int m0 = row0 + wm * 32 + mb * 16 + mbase;   // 8 consecutive rows
#pragma unroll
    for (int t = 0; t < 4; ++t) {
      const v8f a = acc[mb * 4 + t];
      const int colb = col0 + wn * 64 + 16 * t;
      if (colb < 1024) {                // q part: plain f32 row-major store
        const int col = colb + nn;
#pragma unroll
        for (int r = 0; r < 8; ++r)
          QRAW[(size_t)(m0 + r) * 1024 + col] = a[r];
      } else {                          // k/v part: transposed bf16 store
        Pack8 pk;
#pragma unroll
        for (int r = 0; r < 8; ++r) pk.e[r] = (bf16_t)a[r];
        int c = colb + nn - 1024;
        bf16_t* dst = KT;
        if (c >= 1024) { c -= 1024; dst = VT; }
        const int bb   = m0 >> 13;      // batch index (rows per batch = 8192)
        const int nloc = m0 & 8191;     // multiple of 8 -> 16B aligned
        *(uint4*)(dst + ((size_t)(bb * 1024 + c) << 13) + nloc) = pk.u;
      }
    }
  }
}

// ---------------------------------------------------------------------------
// k softmax over sequence dim: one block per (b,h,dd) row of KT (8192 elems).
// ---------------------------------------------------------------------------
__global__ __launch_bounds__(256) void k_col_softmax(const bf16_t* __restrict__ KT,
                                                     bf16_t* __restrict__ KTE,
                                                     float* __restrict__ INVS) {
  __shared__ float red[256];
  const int t = threadIdx.x;
  const size_t row = blockIdx.x;
  const bf16_t* src = KT + row * 8192;
  float v[32];
  float mx = -3.4e38f;
#pragma unroll
  for (int i = 0; i < 32; ++i) {
    v[i] = (float)src[t + (i << 8)];
    mx = fmaxf(mx, v[i]);
  }
  red[t] = mx; __syncthreads();
  for (int o = 128; o > 0; o >>= 1) {
    if (t < o) red[t] = fmaxf(red[t], red[t + o]);
    __syncthreads();
  }
  const float m = red[0];
  __syncthreads();
  float s = 0.f;
#pragma unroll
  for (int i = 0; i < 32; ++i) { v[i] = __expf(v[i] - m); s += v[i]; }
  red[t] = s; __syncthreads();
  for (int o = 128; o > 0; o >>= 1) {
    if (t < o) red[t] += red[t + o];
    __syncthreads();
  }
  bf16_t* dst = KTE + row * 8192;
#pragma unroll
  for (int i = 0; i < 32; ++i) dst[t + (i << 8)] = (bf16_t)v[i];
  if (t == 0) INVS[row] = 1.0f / red[0];
}

// ---------------------------------------------------------------------------
// q softmax over head dim (64) * SCALE; one thread per (row, head).
// ---------------------------------------------------------------------------
__global__ __launch_bounds__(256) void q_row_softmax(const float* __restrict__ QRAW,
                                                     bf16_t* __restrict__ QS) {
  const size_t idx = (size_t)blockIdx.x * 256 + threadIdx.x;   // 32768*16
  const size_t row = idx >> 4;
  const int h = (int)(idx & 15);
  const float4* p = (const float4*)(QRAW + row * 1024 + h * 64);
  float v[64];
#pragma unroll
  for (int i = 0; i < 16; ++i) {
    const float4 f = p[i];
    v[4 * i + 0] = f.x; v[4 * i + 1] = f.y;
    v[4 * i + 2] = f.z; v[4 * i + 3] = f.w;
  }
  float mx = -3.4e38f;
#pragma unroll
  for (int i = 0; i < 64; ++i) mx = fmaxf(mx, v[i]);
  float s = 0.f;
#pragma unroll
  for (int i = 0; i < 64; ++i) { v[i] = __expf(v[i] - mx); s += v[i]; }
  const float r = 0.125f / s;          // SCALE = 64^-0.5 folded in
  bf16_t* o = QS + row * 1024 + h * 64;
#pragma unroll
  for (int g = 0; g < 8; ++g) {
    Pack8 pk;
#pragma unroll
    for (int j = 0; j < 8; ++j) pk.e[j] = (bf16_t)(v[8 * g + j] * r);
    ((uint4*)o)[g] = pk.u;
  }
}

// ---------------------------------------------------------------------------
// GEMM2: context partials.  Per (bh, chunk): D[64x64] += KTE[64 x 1024] * VT^T.
// Deterministic: each chunk writes its own f32 partial slab.
// ---------------------------------------------------------------------------
struct TileCtx { v16bf a, b0, b1; };

__device__ __forceinline__ void load_tileCtx(TileCtx& T,
                                             const bf16_t* __restrict__ A,
                                             const bf16_t* __restrict__ B,
                                             int k) {
  T.a  = load_a_frag(A + k, 8192);
  T.b0 = load_b_frag(B + k, 8192);
  T.b1 = load_b_frag(B + (size_t)16 * 8192 + k, 8192);
}

__global__ __launch_bounds__(256, 1) void gemm_ctx(const bf16_t* __restrict__ KTE,
                                                   const bf16_t* __restrict__ VT,
                                                   float* __restrict__ CTXP) {
  const int bh    = blockIdx.x;        // 0..63
  const int chunk = blockIdx.y;        // 0..7 (K chunks of 1024)
  const int wave  = threadIdx.x >> 5;
  const int lane  = threadIdx.x & 31;
  const int m0 = (wave >> 1) * 16;     // dd tile
  const int n0 = (wave & 1) * 32;      // e tile pair
  const bf16_t* A = KTE + ((size_t)bh << 19) + (size_t)m0 * 8192 + chunk * 1024;
  const bf16_t* B = VT  + ((size_t)bh << 19) + (size_t)n0 * 8192 + chunk * 1024;
  v8f acc0 = zero8(), acc1 = zero8();

  TileCtx T0, T1;
  load_tileCtx(T0, A, B, 0);
  load_tileCtx(T1, A, B, 32);
  for (int k = 0; k < 1024 - 64; k += 64) {
    acc0 = wmma_bf16(T0.a, T0.b0, acc0);
    acc1 = wmma_bf16(T0.a, T0.b1, acc1);
    load_tileCtx(T0, A, B, k + 64);
    acc0 = wmma_bf16(T1.a, T1.b0, acc0);
    acc1 = wmma_bf16(T1.a, T1.b1, acc1);
    load_tileCtx(T1, A, B, k + 96);
  }
  acc0 = wmma_bf16(T0.a, T0.b0, acc0);
  acc1 = wmma_bf16(T0.a, T0.b1, acc1);
  acc0 = wmma_bf16(T1.a, T1.b0, acc0);
  acc1 = wmma_bf16(T1.a, T1.b1, acc1);

  float* out = CTXP + ((size_t)chunk * 64 + bh) * 4096;   // [dd][e]
  const int mbase = (lane >> 4) << 3;
  const int nn    = lane & 15;
#pragma unroll
  for (int r = 0; r < 8; ++r) {
    const int dd = m0 + mbase + r;
    out[dd * 64 + n0 + nn]      = acc0[r];
    out[dd * 64 + n0 + 16 + nn] = acc1[r];
  }
}

// Sum 8 chunk partials, scale rows by 1/colsum, store context TRANSPOSED
// (CTXT[bh][e][dd], bf16) so GEMM3's B operand is K(=dd)-contiguous.
__global__ __launch_bounds__(256) void ctx_finalize(const float* __restrict__ CTXP,
                                                    const float* __restrict__ INVS,
                                                    bf16_t* __restrict__ CTXT) {
  const int idx = blockIdx.x * 256 + threadIdx.x;   // 64*4096
  const int bh = idx >> 12;
  const int dd = (idx >> 6) & 63;
  const int e  = idx & 63;
  float s = 0.f;
#pragma unroll
  for (int c = 0; c < 8; ++c)
    s += CTXP[((size_t)c * 64 + bh) * 4096 + (idx & 4095)];
  CTXT[((size_t)bh << 12) + (e << 6) + dd] = (bf16_t)(s * INVS[(bh << 6) + dd]);
}

// ---------------------------------------------------------------------------
// GEMM3: attn = q_soft @ context per head.  K=64 -> 2 WMMA steps per tile.
// grid.x = h*4 + e-tile, grid.y = 256 (128-row blocks), 8 waves x 16 rows.
// ---------------------------------------------------------------------------
__global__ __launch_bounds__(256, 1) void gemm_attn(const bf16_t* __restrict__ QS,
                                                    const bf16_t* __restrict__ CTXT,
                                                    bf16_t* __restrict__ ATTN) {
  const int h  = blockIdx.x >> 2;
  const int e0 = (blockIdx.x & 3) * 16;
  const int wave = threadIdx.x >> 5;
  const int lane = threadIdx.x & 31;
  const int row0 = blockIdx.y * 128 + wave * 16;
  const int bh = (row0 >> 13) * 16 + h;
  const bf16_t* A = QS + (size_t)row0 * 1024 + h * 64;
  const bf16_t* B = CTXT + ((size_t)bh << 12) + e0 * 64;
  const v16bf a0 = load_a_frag(A, 1024);
  const v16bf b0 = load_b_frag(B, 64);
  const v16bf a1 = load_a_frag(A + 32, 1024);
  const v16bf b1 = load_b_frag(B + 32, 64);
  v8f acc = zero8();
  acc = wmma_bf16(a0, b0, acc);
  acc = wmma_bf16(a1, b1, acc);
  const int mbase = (lane >> 4) << 3;
  const int nn    = lane & 15;
  bf16_t* o = ATTN + (size_t)(row0 + mbase) * 1024 + h * 64 + e0 + nn;
#pragma unroll
  for (int r = 0; r < 8; ++r) o[(size_t)r * 1024] = (bf16_t)acc[r];
}

// ---------------------------------------------------------------------------
// GEMM4: out = attn @ w_out.  M=32768, N=1024, K=1024.  f32 output.
// Same LDS-staged structure as GEMM1.
// ---------------------------------------------------------------------------
__global__ __launch_bounds__(256, 1) void gemm_out(const bf16_t* __restrict__ ATTN,
                                                   const bf16_t* __restrict__ WOUTT,
                                                   float* __restrict__ OUT) {
  __shared__ bf16_t Asl[2][128 * LDS_PAD];
  __shared__ bf16_t Bsl[2][128 * LDS_PAD];
  const int lane = threadIdx.x & 31;
  const int wave = threadIdx.x >> 5;
  const int wm = wave >> 1;
  const int wn = wave & 1;
  const int row0 = blockIdx.y * 128;
  const int col0 = blockIdx.x * 128;
  v8f acc[8];
#pragma unroll
  for (int i = 0; i < 8; ++i) acc[i] = zero8();

  const bf16_t* Ag = ATTN + (size_t)row0 * 1024;
  const bf16_t* Bg = WOUTT + (size_t)col0 * 1024;

  uint4 ra0, ra1, rb0, rb1;
  slab_load(ra0, ra1, Ag, 1024);
  slab_load(rb0, rb1, Bg, 1024);
  slab_store(Asl[0], ra0, ra1);
  slab_store(Bsl[0], rb0, rb1);
  __syncthreads();

  for (int kk = 0; kk < 1024; kk += 32) {
    const int cur = (kk >> 5) & 1;
    const bool more = (kk + 32) < 1024;
    if (more) {
      slab_load(ra0, ra1, Ag + kk + 32, 1024);
      slab_load(rb0, rb1, Bg + kk + 32, 1024);
    }
    const bf16_t* Ab = Asl[cur] + (wm * 32) * LDS_PAD;
    const bf16_t* Bb = Bsl[cur] + (wn * 64) * LDS_PAD;
    const v16bf a0 = load_a_frag(Ab, LDS_PAD);
    const v16bf a1 = load_a_frag(Ab + 16 * LDS_PAD, LDS_PAD);
    v16bf bf[4];
#pragma unroll
    for (int t = 0; t < 4; ++t)
      bf[t] = load_b_frag(Bb + 16 * t * LDS_PAD, LDS_PAD);
#pragma unroll
    for (int t = 0; t < 4; ++t) {
      acc[t]     = wmma_bf16(a0, bf[t], acc[t]);
      acc[4 + t] = wmma_bf16(a1, bf[t], acc[4 + t]);
    }
    if (more) {
      slab_store(Asl[cur ^ 1], ra0, ra1);
      slab_store(Bsl[cur ^ 1], rb0, rb1);
    }
    __syncthreads();
  }

  const int mbase = (lane >> 4) << 3;
  const int nn    = lane & 15;
#pragma unroll
  for (int mb = 0; mb < 2; ++mb) {
    const int m0 = row0 + wm * 32 + mb * 16 + mbase;
#pragma unroll
    for (int t = 0; t < 4; ++t) {
      const v8f a = acc[mb * 4 + t];
#pragma unroll
      for (int r = 0; r < 8; ++r)
        OUT[(size_t)(m0 + r) * 1024 + col0 + wn * 64 + 16 * t + nn] = a[r];
    }
  }
}

// ---------------------------------------------------------------------------
// Launch
// ---------------------------------------------------------------------------
extern "C" void kernel_launch(void* const* d_in, const int* in_sizes, int n_in,
                              void* d_out, int out_size, void* d_ws, size_t ws_size,
                              hipStream_t stream) {
  (void)in_sizes; (void)n_in; (void)out_size; (void)ws_size;
  const float* x     = (const float*)d_in[0];   // [4, 8192, 1024] f32
  const float* w_qkv = (const float*)d_in[1];   // [1024, 3072]    f32
  const float* w_out = (const float*)d_in[2];   // [1024, 1024]    f32
  float* out = (float*)d_out;                   // [4, 8192, 1024] f32

  const size_t M = 32768, D = 1024, N3 = 3072;

  size_t off = 0;
  auto take = [&](size_t bytes) -> char* {
    char* p = (char*)d_ws + off;
    off += (bytes + 255) & ~(size_t)255;
    return p;
  };
  bf16_t* XB    = (bf16_t*)take(M * D * 2);                 //  64 MB
  bf16_t* WQKVT = (bf16_t*)take(N3 * D * 2);                //   6 MB
  bf16_t* WOUTT = (bf16_t*)take(D * D * 2);                 //   2 MB
  float*  QRAW  = (float*) take(M * D * 4);                 // 128 MB
  bf16_t* KT    = (bf16_t*)take((size_t)4096 * 8192 * 2);   //  64 MB
  bf16_t* VT    = (bf16_t*)take((size_t)4096 * 8192 * 2);   //  64 MB
  bf16_t* KTE   = (bf16_t*)take((size_t)4096 * 8192 * 2);   //  64 MB
  float*  CTXP  = (float*) take((size_t)8 * 64 * 4096 * 4); //   8 MB
  bf16_t* CTXT  = (bf16_t*)take((size_t)64 * 4096 * 2);     // 0.5 MB
  float*  INVS  = (float*) take(4096 * 4);
  bf16_t* QS    = KT;   // alias: KT fully consumed by k_col_softmax before this
  bf16_t* ATTN  = XB;   // alias: XB fully consumed by gemm_qkv before this

  cvt_f32_bf16_x4<<<(unsigned)((M * D / 4 + 255) / 256), 256, 0, stream>>>(
      x, XB, (long)(M * D / 4));
  transpose_f32_bf16<<<(unsigned)(((long)N3 * D + 255) / 256), 256, 0, stream>>>(
      w_qkv, WQKVT, (int)D, (int)N3);
  transpose_f32_bf16<<<(unsigned)(((long)D * D + 255) / 256), 256, 0, stream>>>(
      w_out, WOUTT, (int)D, (int)D);
  gemm_qkv   <<<dim3(24, 256), 256, 0, stream>>>(XB, WQKVT, QRAW, KT, VT);
  k_col_softmax<<<4096, 256, 0, stream>>>(KT, KTE, INVS);
  q_row_softmax<<<2048, 256, 0, stream>>>(QRAW, QS);
  gemm_ctx   <<<dim3(64, 8), 256, 0, stream>>>(KTE, VT, CTXP);
  ctx_finalize<<<1024, 256, 0, stream>>>(CTXP, INVS, CTXT);
  gemm_attn  <<<dim3(64, 256), 256, 0, stream>>>(QS, CTXT, ATTN);
  gemm_out   <<<dim3(8, 256), 256, 0, stream>>>(ATTN, WOUTT, out);
}
